// Zoomer_34359738368366
// MI455X (gfx1250) — matrix operand order repaired and checked by reading, not compiled, not run on hardware
//
#include <hip/hip_runtime.h>

// ---------------------------------------------------------------------------
// HAN-style graph attention for MI455X (gfx1250, wave32, WMMA).
//   - prep kernel: fp32 weights -> bf16 W^T in d_ws (L2-resident afterwards)
//   - main kernel: one block per (batch, node-type); 8 waves; GAT k/v
//     projections via v_wmma_f32_16x16x32_bf16; attention/softmax/sem layer
//     via VALU in LDS.
// ---------------------------------------------------------------------------

#define EMB 64
#define SEQ 50
#define SPAD 64            // padded sequence (4 M-tiles of 16)
#define NBR_STRIDE 72      // ushort elements per LDS neighbor row (bank spread)
#define KV_STRIDE 68       // f32 elements per LDS kp/vp row (bank spread)

typedef __attribute__((ext_vector_type(16))) __bf16          v16bf;
typedef __attribute__((ext_vector_type(8)))  float           v8f;
typedef __attribute__((ext_vector_type(16))) unsigned short  u16x16;
typedef __attribute__((ext_vector_type(8)))  unsigned short  u16x8;
typedef __attribute__((ext_vector_type(4)))  float           f32x4;

__device__ __forceinline__ unsigned short f32_to_bf16(float f) {
  unsigned u = __builtin_bit_cast(unsigned, f);
  u += 0x7FFFu + ((u >> 16) & 1u);       // round to nearest even
  return (unsigned short)(u >> 16);
}
__device__ __forceinline__ float bf16_to_f32(unsigned short h) {
  unsigned u = ((unsigned)h) << 16;
  return __builtin_bit_cast(float, u);
}

// ---------------------------------------------------------------------------
// Prep: wsT[set(2)][t(3)][mat(4)][n(64)][k(64)] (bf16) = W[set][t][mat][k][n]
// set 0 = gat_W, set 1 = sem_W.  Transposed so B-fragment rows (per output
// column n) are K-contiguous.
// ---------------------------------------------------------------------------
__global__ void prep_weights_kernel(const float* __restrict__ gat_W,
                                    const float* __restrict__ sem_W,
                                    unsigned short* __restrict__ wsT) {
  int i = blockIdx.x * blockDim.x + threadIdx.x;
  const int total = 2 * 3 * 4 * 64 * 64;
  if (i >= total) return;
  int k   = i & 63;
  int n   = (i >> 6) & 63;
  int mat = i >> 12;                       // 0..23
  const float* src = (mat < 12) ? gat_W : sem_W;
  int m = (mat < 12) ? mat : (mat - 12);   // t*4 + j
  wsT[i] = f32_to_bf16(src[m * 4096 + k * 64 + n]);
}

// ---------------------------------------------------------------------------
// Main kernel: blockIdx.x = batch, blockIdx.y = node type (0=user,1=movie,2=query)
// ---------------------------------------------------------------------------
__global__ void __launch_bounds__(256)
han_kernel(const int* __restrict__ uids, const int* __restrict__ qids,
           const int* __restrict__ mids,
           const int* __restrict__ u_movies, const int* __restrict__ m_querys,
           const int* __restrict__ m_users,  const int* __restrict__ q_movies,
           const float* __restrict__ user_table,
           const float* __restrict__ item_table,
           const float* __restrict__ query_table,
           const float* __restrict__ gat_b, const float* __restrict__ sem_b,
           const unsigned short* __restrict__ wT,
           float* __restrict__ out) {
  __shared__ unsigned short sh_nbr[SPAD * NBR_STRIDE];  // bf16 neighbor tile
  __shared__ float sh_kp[SPAD * KV_STRIDE];             // relu(K @ Wk + bk)
  __shared__ float sh_vp[SPAD * KV_STRIDE];             // relu(V @ Wv + bv)
  __shared__ float sh_node[EMB];
  __shared__ float sh_q[EMB];
  __shared__ float sh_sc[SPAD];
  __shared__ float sh_a[SPAD];
  __shared__ float sh_y[EMB];
  __shared__ float sh_key[3 * EMB];                     // node, agg0, agg1
  __shared__ float sh_kps[3 * EMB];
  __shared__ float sh_vps[3 * EMB];

  const int b   = blockIdx.x;
  const int t   = blockIdx.y;
  const int tid = threadIdx.x;

  const float* node_table; const int* node_ids;
  const int* nbr_idx0; const float* nbr_tab0;
  const int* nbr_idx1 = nullptr; const float* nbr_tab1 = nullptr;
  if (t == 0)      { node_table = user_table;  node_ids = uids;
                     nbr_idx0 = u_movies; nbr_tab0 = item_table; }
  else if (t == 1) { node_table = item_table; node_ids = mids;
                     nbr_idx0 = m_querys; nbr_tab0 = query_table;
                     nbr_idx1 = m_users;  nbr_tab1 = user_table; }
  else             { node_table = query_table; node_ids = qids;
                     nbr_idx0 = q_movies; nbr_tab0 = item_table; }
  const int nsets = (t == 1) ? 2 : 1;
  const int L     = 1 + nsets;

  const unsigned short* gT = wT + (size_t)(t * 4) * 4096;        // gat W^T (bf16)
  const unsigned short* sT = wT + (size_t)(12 + t * 4) * 4096;   // sem W^T (bf16)
  const float* gb = gat_b + t * 4 * 64;
  const float* sb = sem_b + t * 4 * 64;

  if (tid < EMB) {
    int nid = node_ids[b];
    float v = node_table[(size_t)nid * EMB + tid];
    sh_node[tid] = v;
    sh_key[tid]  = v;
  }

  for (int set = 0; set < nsets; ++set) {
    const int*   nidx = set ? nbr_idx1 : nbr_idx0;
    const float* ntab = set ? nbr_tab1 : nbr_tab0;

    // ---- Phase A: gather neighbors -> bf16 LDS tile (pad rows zeroed) ----
    {
      int row  = tid >> 2;          // 0..63
      int coff = (tid & 3) * 16;    // 16 elements per thread
      unsigned short tmp[16];
      if (row < SEQ) {
        int idx = nidx[b * SEQ + row];
        const f32x4* src = (const f32x4*)(ntab + (size_t)idx * EMB + coff);
        f32x4 d0 = src[0], d1 = src[1], d2 = src[2], d3 = src[3];
        #pragma unroll
        for (int e = 0; e < 4; ++e) {
          tmp[e]      = f32_to_bf16(d0[e]);
          tmp[e + 4]  = f32_to_bf16(d1[e]);
          tmp[e + 8]  = f32_to_bf16(d2[e]);
          tmp[e + 12] = f32_to_bf16(d3[e]);
        }
      } else {
        #pragma unroll
        for (int e = 0; e < 16; ++e) tmp[e] = 0;
      }
      u16x8 lo, hi;
      #pragma unroll
      for (int e = 0; e < 8; ++e) { lo[e] = tmp[e]; hi[e] = tmp[e + 8]; }
      *(u16x8*)(&sh_nbr[row * NBR_STRIDE + coff])     = lo;
      *(u16x8*)(&sh_nbr[row * NBR_STRIDE + coff + 8]) = hi;
    }
    __syncthreads();

    // ---- Phase B: WMMA projections  kp/vp = relu(N @ W{k,v} + b) ----
    {
      const int w    = tid >> 5;     // wave 0..7
      const int lane = tid & 31;
      const int p    = w >> 2;       // 0 -> k-proj, 1 -> v-proj
      const int n    = w & 3;        // output column tile
      const int jmat = p ? 2 : 1;    // W[1]=k, W[2]=v
      const unsigned short* WTm = gT + (size_t)jmat * 4096;
      const float bv = gb[jmat * 64 + n * 16 + (lane & 15)];
      float* dstLDS = p ? sh_vp : sh_kp;

      // B fragments (K x N loaded from W^T, K-contiguous per lane):
      //   lane 0-15: N=lane,    K = kb*32 + 0..15
      //   lane16-31: N=lane-16, K = kb*32 + 16..31
      v16bf bfrag[2];
      #pragma unroll
      for (int kb = 0; kb < 2; ++kb) {
        int ncol = n * 16 + (lane & 15);
        int kk   = kb * 32 + (lane >> 4) * 16;
        bfrag[kb] = __builtin_bit_cast(v16bf,
                      *(const u16x16*)(WTm + ncol * 64 + kk));
      }

      #pragma unroll
      for (int m = 0; m < 4; ++m) {
        v8f acc;
        #pragma unroll
        for (int e = 0; e < 8; ++e) acc[e] = bv;   // bias per output column
        #pragma unroll
        for (int kb = 0; kb < 2; ++kb) {
          // A fragment (16x32 bf16): lane 0-15 -> K kb*32+0..15,
          //                          lane16-31 -> K kb*32+16..31
          int row = m * 16 + (lane & 15);
          int kk  = kb * 32 + (lane >> 4) * 16;
          u16x8 alo = *(const u16x8*)(&sh_nbr[row * NBR_STRIDE + kk]);
          u16x8 ahi = *(const u16x8*)(&sh_nbr[row * NBR_STRIDE + kk + 8]);
          u16x16 av;
          #pragma unroll
          for (int e = 0; e < 8; ++e) { av[e] = alo[e]; av[e + 8] = ahi[e]; }
          v16bf afrag = __builtin_bit_cast(v16bf, av);
          acc = __builtin_amdgcn_wmma_f32_16x16x32_bf16(
                    false, afrag, false, bfrag[kb], (short)0, acc, false, false);
        }
        #pragma unroll
        for (int v2 = 0; v2 < 8; ++v2) {          // C/D layout: VGPR v -> M=v(+8)
          int row = m * 16 + v2 + 8 * (lane >> 4);
          int col = n * 16 + (lane & 15);
          float val = acc[v2];
          dstLDS[row * KV_STRIDE + col] = val > 0.f ? val : 0.f;
        }
      }
    }
    __syncthreads();

    // ---- Phase C: qp = relu(node @ Wq + bq) ----
    if (tid < EMB) {
      const unsigned short* wr = gT + tid * 64;          // W[0]^T row tid
      float s = gb[tid];
      for (int k = 0; k < EMB; ++k) s += sh_node[k] * bf16_to_f32(wr[k]);
      sh_q[tid] = s > 0.f ? s : 0.f;
    }
    __syncthreads();

    // ---- Phase D: scores = qp . kp_s / 8 ----
    if (tid < SPAD) {
      float s = -1e30f;
      if (tid < SEQ) {
        s = 0.f;
        for (int k = 0; k < EMB; ++k) s += sh_q[k] * sh_kp[tid * KV_STRIDE + k];
        s *= 0.125f;
      }
      sh_sc[tid] = s;
    }
    __syncthreads();

    // ---- Phase E: softmax numerators ----
    if (tid < SPAD) {
      float m = -1e30f;
      for (int s2 = 0; s2 < SPAD; ++s2) m = fmaxf(m, sh_sc[s2]);
      sh_a[tid] = expf(sh_sc[tid] - m);                  // pads -> exp(-inf)=0
    }
    __syncthreads();

    // ---- Phase F: y = softmax @ vp ----
    if (tid < EMB) {
      float denom = 0.f;
      for (int s2 = 0; s2 < SPAD; ++s2) denom += sh_a[s2];
      float y = 0.f;
      for (int s2 = 0; s2 < SEQ; ++s2) y += sh_a[s2] * sh_vp[s2 * KV_STRIDE + tid];
      sh_y[tid] = y / denom;
    }
    __syncthreads();

    // ---- Phase G: agg = relu(y @ W3 + b3) -> key row ----
    if (tid < EMB) {
      const unsigned short* wr = gT + 3 * 4096 + tid * 64;
      float s = gb[3 * 64 + tid];
      for (int k = 0; k < EMB; ++k) s += sh_y[k] * bf16_to_f32(wr[k]);
      sh_key[(1 + set) * EMB + tid] = s > 0.f ? s : 0.f;
    }
    __syncthreads();
  }

  // ======== Semantic attention over L in {2,3} key rows (tiny, VALU) ========
  if (tid < L * EMB) {                       // kp_s / vp_s
    int r = tid >> 6, j = tid & 63;
    const float* krow = sh_key + r * EMB;
    const unsigned short* Wk = sT + 1 * 4096 + j * 64;
    const unsigned short* Wv = sT + 2 * 4096 + j * 64;
    float sk = sb[1 * 64 + j], sv = sb[2 * 64 + j];
    for (int k = 0; k < EMB; ++k) {
      float x = krow[k];
      sk += x * bf16_to_f32(Wk[k]);
      sv += x * bf16_to_f32(Wv[k]);
    }
    sh_kps[tid] = sk > 0.f ? sk : 0.f;
    sh_vps[tid] = sv > 0.f ? sv : 0.f;
  }
  if (tid >= 192) {                          // qp_s (threads 192..255)
    int j = tid - 192;
    const unsigned short* Wq = sT + j * 64;
    float s = sb[j];
    for (int k = 0; k < EMB; ++k) s += sh_node[k] * bf16_to_f32(Wq[k]);
    sh_q[j] = s > 0.f ? s : 0.f;
  }
  __syncthreads();

  if (tid < L) {                             // scores
    float s = 0.f;
    for (int k = 0; k < EMB; ++k) s += sh_q[k] * sh_kps[tid * EMB + k];
    sh_sc[tid] = s * 0.125f;
  }
  __syncthreads();

  if (tid < EMB) {                           // softmax + y
    float m = -1e30f;
    for (int r = 0; r < L; ++r) m = fmaxf(m, sh_sc[r]);
    float denom = 0.f, y = 0.f;
    for (int r = 0; r < L; ++r) {
      float a = expf(sh_sc[r] - m);
      denom += a;
      y += a * sh_vps[r * EMB + tid];
    }
    sh_y[tid] = y / denom;
  }
  __syncthreads();

  if (tid < EMB) {                           // out = relu(y @ W3_s + b3_s)
    const unsigned short* wr = sT + 3 * 4096 + tid * 64;
    float s = sb[3 * 64 + tid];
    for (int k = 0; k < EMB; ++k) s += sh_y[k] * bf16_to_f32(wr[k]);
    out[(size_t)b * 192 + t * 64 + tid] = s > 0.f ? s : 0.f;
  }
}

// ---------------------------------------------------------------------------
extern "C" void kernel_launch(void* const* d_in, const int* in_sizes, int n_in,
                              void* d_out, int out_size, void* d_ws, size_t ws_size,
                              hipStream_t stream) {
  (void)n_in; (void)out_size; (void)ws_size;
  const int*   uids        = (const int*)d_in[0];
  const int*   qids        = (const int*)d_in[1];
  const int*   mids        = (const int*)d_in[2];
  const int*   u_movies    = (const int*)d_in[3];
  const int*   m_querys    = (const int*)d_in[4];
  const int*   m_users     = (const int*)d_in[5];
  const int*   q_movies    = (const int*)d_in[6];
  const float* user_table  = (const float*)d_in[7];
  const float* item_table  = (const float*)d_in[8];
  const float* query_table = (const float*)d_in[9];
  const float* gat_W       = (const float*)d_in[10];
  const float* gat_b       = (const float*)d_in[11];
  const float* sem_W       = (const float*)d_in[12];
  const float* sem_b       = (const float*)d_in[13];
  float* out = (float*)d_out;
  unsigned short* wT = (unsigned short*)d_ws;   // 2*3*4*4096 bf16 = 192 KB

  const int B = in_sizes[0];
  const int wtotal = 2 * 3 * 4 * 64 * 64;
  prep_weights_kernel<<<(wtotal + 255) / 256, 256, 0, stream>>>(gat_W, sem_W, wT);

  dim3 grid(B, 3);
  han_kernel<<<grid, 256, 0, stream>>>(uids, qids, mids,
                                       u_movies, m_querys, m_users, q_movies,
                                       user_table, item_table, query_table,
                                       gat_b, sem_b, wT, out);
}
